// MonetRouter_88433376625148
// MI455X (gfx1250) — matrix operand order repaired and checked by reading, not compiled, not run on hardware
//
#include <hip/hip_runtime.h>
#include <hip/hip_bf16.h>

// ---------------- problem constants ----------------
#define HEADS   8
#define EXPERTS 512
#define MROWS   4096            // B*S = 2*2048
#define KDIM    2048            // D
#define NDIM    4096            // HEADS*EXPERTS
#define TENSOR_ELEMS 16777216   // MROWS*NDIM

#define TILE_M 256              // block tile M (4 waves x 64)
#define TILE_N 128              // block tile N (2 waves x 64)
#define TILE_K 32
#define LDA    40               // padded LDS row stride (elements)

#define SIGMA_F  2.153864f      // norm.ppf(1 - 8/512)
#define BN_EPS   1e-5

typedef __attribute__((ext_vector_type(16))) __bf16 bf16x16;
typedef __attribute__((ext_vector_type(8)))  __bf16 bf16x8;
typedef __attribute__((ext_vector_type(4)))  __bf16 bf16x4;
typedef __attribute__((ext_vector_type(8)))  float  f32x8;
typedef __attribute__((ext_vector_type(4)))  float  f32x4;

union FragBF { bf16x16 v; bf16x8 h[2]; };

#define WMMA_BF16(A, B, C) \
  __builtin_amdgcn_wmma_f32_16x16x32_bf16(false, (A), false, (B), (short)0, (C), false, false)

__device__ inline void split4(const f32x4 f, bf16x4& hi, bf16x4& lo)
{
#pragma unroll
    for (int i = 0; i < 4; ++i) {
        float  a = f[i];
        __bf16 h = (__bf16)a;
        hi[i] = h;
        lo[i] = (__bf16)(a - (float)h);
    }
}

// =====================================================================
// Phase 1: C[g] = X @ W[g]^T with bf16x3 split-precision WMMA
// grid = (MROWS/256, NDIM/128, 2), block = 256 (8 waves, 4(M) x 2(N))
// wave tile 64x64: 16 accum tiles, 48 WMMA : 32 ds_load per K-step
// =====================================================================
__global__ __launch_bounds__(256)
void monet_gemm(const float* __restrict__ X, const float* __restrict__ W1,
                const float* __restrict__ W2, float* __restrict__ Out)
{
    __shared__ __bf16 sAhi[TILE_M * LDA];   // 20 KB
    __shared__ __bf16 sAlo[TILE_M * LDA];   // 20 KB
    __shared__ __bf16 sBhi[TILE_N * LDA];   // 10 KB
    __shared__ __bf16 sBlo[TILE_N * LDA];   // 10 KB

    const int g = blockIdx.z;
    const float* __restrict__ W = g ? W2 : W1;
    float* __restrict__ C = Out + (size_t)g * (size_t)TENSOR_ELEMS;

    const int bm   = blockIdx.x * TILE_M;
    const int bn   = blockIdx.y * TILE_N;
    const int tid  = threadIdx.x;
    const int lane = tid & 31;
    const int wave = tid >> 5;
    const int wm   = wave & 3;       // wave M slot: rows wm*64
    const int wn   = wave >> 2;      // wave N slot: cols wn*64

    // staging map: lane group of 8 covers 128 contiguous bytes of one row
    const int srow  = tid >> 3;          // 0..31
    const int scol4 = (tid & 7) << 2;    // 0,4,...,28

    const float* gA = X + (size_t)(bm + srow) * KDIM + scol4;
    const float* gB = W + (size_t)(bn + srow) * KDIM + scol4;

    f32x8 acc[4][4];
#pragma unroll
    for (int i = 0; i < 4; ++i)
#pragma unroll
        for (int j = 0; j < 4; ++j)
#pragma unroll
            for (int e = 0; e < 8; ++e) acc[i][j][e] = 0.0f;

    for (int k0 = 0; k0 < KDIM; k0 += TILE_K) {
        // ---- global loads: A 8 chunks (256 rows), B 4 chunks (128 rows) ----
        f32x4 av[8], bv[4];
#pragma unroll
        for (int c = 0; c < 8; ++c)
            av[c] = *(const f32x4*)(gA + (size_t)(32 * c) * KDIM);
#pragma unroll
        for (int c = 0; c < 4; ++c)
            bv[c] = *(const f32x4*)(gB + (size_t)(32 * c) * KDIM);
        __builtin_prefetch(gA + TILE_K, 0, 1);   // global_prefetch next K tile
        __builtin_prefetch(gB + TILE_K, 0, 1);

        // ---- split fp32 -> bf16 hi + lo ----
        bf16x4 ahi[8], alo[8], bhi[4], blo[4];
#pragma unroll
        for (int c = 0; c < 8; ++c) split4(av[c], ahi[c], alo[c]);
#pragma unroll
        for (int c = 0; c < 4; ++c) split4(bv[c], bhi[c], blo[c]);

        __syncthreads();   // previous iteration's fragment reads done
#pragma unroll
        for (int c = 0; c < 8; ++c) {
            *(bf16x4*)&sAhi[(srow + 32 * c) * LDA + scol4] = ahi[c];
            *(bf16x4*)&sAlo[(srow + 32 * c) * LDA + scol4] = alo[c];
        }
#pragma unroll
        for (int c = 0; c < 4; ++c) {
            *(bf16x4*)&sBhi[(srow + 32 * c) * LDA + scol4] = bhi[c];
            *(bf16x4*)&sBlo[(srow + 32 * c) * LDA + scol4] = blo[c];
        }
        __syncthreads();

        // ---- build fragments from LDS ----
        const int am   = wm * 64;
        const int bnn  = wn * 64;
        const int r16  = lane & 15;
        const int aoff = (lane >> 4) << 3;   // A: lanes 16-31 hold K=8..15 / 24..31
        const int boff = (lane >> 4) << 3;   // B: lanes 16-31 hold K=8..15 / 24..31

        FragBF Ah[4], Al[4];
#pragma unroll
        for (int i = 0; i < 4; ++i) {
            const __bf16* p = &sAhi[(am + i * 16 + r16) * LDA + aoff];
            Ah[i].h[0] = *(const bf16x8*)(p);
            Ah[i].h[1] = *(const bf16x8*)(p + 16);
            const __bf16* q = &sAlo[(am + i * 16 + r16) * LDA + aoff];
            Al[i].h[0] = *(const bf16x8*)(q);
            Al[i].h[1] = *(const bf16x8*)(q + 16);
        }
        FragBF Bh[4], Bl[4];
#pragma unroll
        for (int j = 0; j < 4; ++j) {
            const __bf16* p = &sBhi[(bnn + j * 16 + r16) * LDA + boff];
            Bh[j].h[0] = *(const bf16x8*)(p);
            Bh[j].h[1] = *(const bf16x8*)(p + 16);
            const __bf16* q = &sBlo[(bnn + j * 16 + r16) * LDA + boff];
            Bl[j].h[0] = *(const bf16x8*)(q);
            Bl[j].h[1] = *(const bf16x8*)(q + 16);
        }

        // ---- bf16x3: hi*hi + lo*hi + hi*lo ----
#pragma unroll
        for (int i = 0; i < 4; ++i)
#pragma unroll
            for (int j = 0; j < 4; ++j) {
                acc[i][j] = WMMA_BF16(Ah[i].v, Bh[j].v, acc[i][j]);
                acc[i][j] = WMMA_BF16(Al[i].v, Bh[j].v, acc[i][j]);
                acc[i][j] = WMMA_BF16(Ah[i].v, Bl[j].v, acc[i][j]);
            }

        gA += TILE_K;
        gB += TILE_K;
    }

    // ---- store C (VGPR v: lanes 0-15 -> M=v, lanes 16-31 -> M=v+8; N=lane%16) ----
    const int rsub = (lane >> 4) << 3;
    const int csub = lane & 15;
#pragma unroll
    for (int i = 0; i < 4; ++i)
#pragma unroll
        for (int j = 0; j < 4; ++j) {
            size_t base = (size_t)(bm + wm * 64 + i * 16 + rsub) * NDIM
                        + (size_t)(bn + wn * 64 + j * 16 + csub);
#pragma unroll
            for (int v = 0; v < 8; ++v)
                C[base + (size_t)v * NDIM] = acc[i][j][v];
        }
}

// =====================================================================
// Phase 2a: partial sums per (tensor, head, row-chunk), fp64, no atomics
// grid = (32 chunks, 8 heads, 2 tensors), block = 256
// =====================================================================
__global__ __launch_bounds__(256)
void monet_stats1(const float* __restrict__ Out, double* __restrict__ partial)
{
    const int chunk = blockIdx.x;
    const int h     = blockIdx.y;
    const int t     = blockIdx.z;
    const float* base = Out + (size_t)t * TENSOR_ELEMS
                            + (size_t)chunk * 128 * NDIM + h * EXPERTS;
    double s = 0.0, ss = 0.0;
    for (int idx = threadIdx.x; idx < 128 * EXPERTS; idx += 256) {
        int r = idx >> 9;
        int e = idx & (EXPERTS - 1);
        float v = base[(size_t)r * NDIM + e];
        s  += (double)v;
        ss += (double)v * (double)v;
    }
    __shared__ double red[512];
    red[threadIdx.x]       = s;
    red[256 + threadIdx.x] = ss;
    __syncthreads();
    for (int off = 128; off > 0; off >>= 1) {
        if (threadIdx.x < off) {
            red[threadIdx.x]       += red[threadIdx.x + off];
            red[256 + threadIdx.x] += red[256 + threadIdx.x + off];
        }
        __syncthreads();
    }
    if (threadIdx.x == 0) {
        int gidx = (t * HEADS + h) * 32 + chunk;
        partial[2 * gidx]     = red[0];
        partial[2 * gidx + 1] = red[256];
    }
}

// =====================================================================
// Phase 2b: finalize mean / inv_std per (tensor, head)
// grid = 16 blocks of one wave (32 threads)
// =====================================================================
__global__ void monet_stats2(const double* __restrict__ partial, float* __restrict__ stats)
{
    const int gidx = blockIdx.x;            // 0..15 = t*8 + h
    double s  = partial[2 * (gidx * 32 + threadIdx.x)];
    double ss = partial[2 * (gidx * 32 + threadIdx.x) + 1];
    for (int off = 16; off > 0; off >>= 1) {
        s  += __shfl_xor(s,  off);
        ss += __shfl_xor(ss, off);
    }
    if (threadIdx.x == 0) {
        const double N = (double)MROWS * (double)EXPERTS;   // 2M per head
        double mean = s / N;
        double var  = ss / N - mean * mean;                 // biased, matches jnp.var
        stats[2 * gidx]     = (float)mean;
        stats[2 * gidx + 1] = (float)(1.0 / sqrt(var + BN_EPS));
    }
}

// =====================================================================
// Phase 3: threshold + masked softmax over 512 experts, in place
// grid = (MROWS*HEADS, 2), block = 256 (2 experts/thread)
// =====================================================================
__global__ __launch_bounds__(256)
void monet_gate(float* __restrict__ Out, const float* __restrict__ stats)
{
    const int r   = blockIdx.x;             // 0..32767, head = r & 7
    const int t   = blockIdx.y;
    const int h   = r & (HEADS - 1);
    const int tid = threadIdx.x;
    const float mean = stats[2 * (t * HEADS + h)];
    const float inv  = stats[2 * (t * HEADS + h) + 1];

    float* row = Out + (size_t)t * TENSOR_ELEMS + (size_t)r * EXPERTS;
    float gz0 = row[tid];
    float gz1 = row[tid + 256];
    float gn0 = (gz0 - mean) * inv;
    float gn1 = (gz1 - mean) * inv;

    __shared__ float red[256];

    // 1) max(gn) -> gs
    red[tid] = fmaxf(gn0, gn1);
    __syncthreads();
    for (int off = 128; off > 0; off >>= 1) {
        if (tid < off) red[tid] = fmaxf(red[tid], red[tid + off]);
        __syncthreads();
    }
    float gs = fminf(red[0], SIGMA_F);
    __syncthreads();

    bool k0 = gn0 >= gs;
    bool k1 = gn1 >= gs;

    // 2) max of kept logits (at least the gn-argmax survives)
    red[tid] = fmaxf(k0 ? gz0 : -3.0e38f, k1 ? gz1 : -3.0e38f);
    __syncthreads();
    for (int off = 128; off > 0; off >>= 1) {
        if (tid < off) red[tid] = fmaxf(red[tid], red[tid + off]);
        __syncthreads();
    }
    float lmax = red[0];
    __syncthreads();

    // 3) exp-sum
    float p0 = k0 ? expf(gz0 - lmax) : 0.0f;
    float p1 = k1 ? expf(gz1 - lmax) : 0.0f;
    red[tid] = p0 + p1;
    __syncthreads();
    for (int off = 128; off > 0; off >>= 1) {
        if (tid < off) red[tid] += red[tid + off];
        __syncthreads();
    }
    float rinv = 1.0f / red[0];

    row[tid]       = p0 * rinv;
    row[tid + 256] = p1 * rinv;
}

// =====================================================================
extern "C" void kernel_launch(void* const* d_in, const int* in_sizes, int n_in,
                              void* d_out, int out_size, void* d_ws, size_t ws_size,
                              hipStream_t stream)
{
    (void)in_sizes; (void)n_in; (void)out_size; (void)ws_size;
    const float* x  = (const float*)d_in[0];
    const float* w1 = (const float*)d_in[1];
    const float* w2 = (const float*)d_in[2];
    float*  out     = (float*)d_out;
    float*  stats   = (float*)d_ws;                       // 32 floats
    double* partial = (double*)((char*)d_ws + 256);       // 1024 doubles

    dim3 gGemm(MROWS / TILE_M, NDIM / TILE_N, 2);
    monet_gemm<<<gGemm, 256, 0, stream>>>(x, w1, w2, out);

    dim3 gS1(32, HEADS, 2);
    monet_stats1<<<gS1, 256, 0, stream>>>(out, partial);
    monet_stats2<<<16, 32, 0, stream>>>(partial, stats);

    dim3 gGate(MROWS * HEADS, 2);
    monet_gate<<<gGate, 256, 0, stream>>>(out, stats);
}
